// BiLSTM_CRF_44367012168007
// MI455X (gfx1250) — compile-verified, matrix operand order checked by reading, not run on hardware
//
#include <hip/hip_runtime.h>
#include <hip/hip_bf16.h>

typedef __bf16 bf16_t;
typedef __attribute__((ext_vector_type(16))) __bf16 v16bf;
typedef __attribute__((ext_vector_type(8)))  float  v8f;
typedef __attribute__((ext_vector_type(4)))  int    v4i_;

// address-space-qualified pointer types for the async-to-LDS builtin
typedef __attribute__((address_space(1))) v4i_* as1_v4i_ptr;   // global
typedef __attribute__((address_space(3))) v4i_* as3_v4i_ptr;   // LDS

#define TO_AS1(p) ((as1_v4i_ptr)(unsigned long long)(p))
#define TO_AS3(p) ((as3_v4i_ptr)(unsigned int)(unsigned long long)(p))

#define V_   50000
#define E_   256
#define H2_  256
#define HID_ 512
#define G4_  1024   // 4*H2
#define K_   17
#define T_   256
#define B_   64

#if defined(__has_builtin)
#if __has_builtin(__builtin_amdgcn_global_load_async_to_lds_b128)
#define HAVE_ASYNC_LDS 1
#endif
#endif
#ifndef HAVE_ASYNC_LDS
#define HAVE_ASYNC_LDS 0
#endif

__device__ __forceinline__ void wait_async0_() {
#if defined(__has_builtin)
#if __has_builtin(__builtin_amdgcn_s_wait_asynccnt)
    __builtin_amdgcn_s_wait_asynccnt(0);
    return;
#else
    asm volatile("s_wait_asynccnt 0" ::: "memory");
    return;
#endif
#else
    asm volatile("s_wait_asynccnt 0" ::: "memory");
#endif
}

// ---------------------------------------------------------------- helpers
__device__ __forceinline__ float sigmoidf_(float x) {
    return 1.0f / (1.0f + __expf(-x));
}
__device__ __forceinline__ float tanhf_(float x) {
    // tanh(x) = 1 - 2/(e^{2x}+1); saturates correctly at +-1
    return 1.0f - 2.0f / (__expf(2.0f * x) + 1.0f);
}

// ------------------------------------------------- embedding gather -> bf16
__global__ void k_embed(const int* __restrict__ sent,
                        const float* __restrict__ emb,
                        bf16_t* __restrict__ xbf) {
    int row = blockIdx.x;            // b*T + t  (16384 rows)
    int e   = threadIdx.x;           // 0..255
    int tok = sent[row];
    float v = (tok == 0) ? 0.0f : emb[(long)tok * E_ + e];
    xbf[(long)row * E_ + e] = (bf16_t)v;
}

// ------------------------------------------------- f32 -> bf16 weight convert
__global__ void k_cvt_bf16(const float* __restrict__ in,
                           bf16_t* __restrict__ out, int n) {
    int i = blockIdx.x * 256 + threadIdx.x;
    if (i < n) out[i] = (bf16_t)in[i];
}

// ------------------------------------------------- gx = x @ W_ih^T + b_ih + b_hh
// A m-tile (16x256 bf16 = 8KB) is shared by all 8 waves of the block: stage it
// once into LDS via the CDNA5 async-to-LDS path, then each wave streams its
// own B tile from global while WMMA-ing.
// out gx: (T, B, G4) f32 -- time-major so each LSTM step reads one contiguous
// 64x1024 block.
__global__ __launch_bounds__(256)
void k_gx(const bf16_t* __restrict__ xbf,
          const bf16_t* __restrict__ wbf,
          const float* __restrict__ b_ih,
          const float* __restrict__ b_hh,
          float* __restrict__ gx) {
    __shared__ __align__(16) bf16_t atile[16 * E_];   // 8 KB

    const int wave  = threadIdx.x >> 5;
    const int lane  = threadIdx.x & 31;
    const int lh    = lane >> 4;       // 0/1 : K-half selector
    const int lm    = lane & 15;
    const int mt    = blockIdx.x >> 3;         // 0..1023
    const int ng    = blockIdx.x & 7;
    const int nt    = ng * 8 + wave;           // 0..63
    const int m0    = mt * 16;
    const int n0    = nt * 16;

    // ---- stage A tile (16 x 256 bf16) into LDS
    const bf16_t* asrc = xbf + (long)m0 * E_;
#if HAVE_ASYNC_LDS
    {
        const int chunk = threadIdx.x * 2;            // 512 x 16B chunks
        __builtin_amdgcn_global_load_async_to_lds_b128(
            TO_AS1(asrc + chunk * 8), TO_AS3(atile + chunk * 8), 0, 0);
        __builtin_amdgcn_global_load_async_to_lds_b128(
            TO_AS1(asrc + (chunk + 1) * 8), TO_AS3(atile + (chunk + 1) * 8), 0, 0);
        wait_async0_();
    }
#else
    for (int i = threadIdx.x; i < 16 * E_; i += 256) atile[i] = asrc[i];
#endif
    __syncthreads();

    const bf16_t* bptr = wbf + (long)(n0 + lm) * E_ + lh * 16;

    v8f acc = {};
#pragma unroll
    for (int k = 0; k < E_; k += 32) {
        v16bf a = *(const v16bf*)&atile[lm * E_ + k + lh * 16];
        v16bf b = *(const v16bf*)(bptr + k);
        acc = __builtin_amdgcn_wmma_f32_16x16x32_bf16(
                  false, a, false, b, (short)0, acc, false, false);
    }

    const int n    = n0 + lm;
    const float bi = b_ih[n] + b_hh[n];
#pragma unroll
    for (int v = 0; v < 8; ++v) {
        int row = m0 + v + 8 * lh;             // row = b*T + t
        int b_  = row >> 8;                    // /T_
        int t_  = row & (T_ - 1);
        gx[((long)(t_ * B_ + b_) << 10) + n] = acc[v] + bi;
    }
}

// ------------------------------------------------- bidirectional LSTM scan
// One workgroup (32 waves) per direction.  h (64x256 bf16 = 32KB) in LDS,
// cell state c stays in each wave's VGPRs across all 256 steps.
// h@W_hh^T accumulates from ZERO so the gx loads (issued at step top) overlap
// the whole WMMA chain; gx is folded in during the gate math.  The next
// step's gx block is prefetched (global_prefetch_b8) while this step runs.
__global__ __launch_bounds__(1024)
void k_lstm(const float*  __restrict__ gx_f, const float*  __restrict__ gx_b,
            const bf16_t* __restrict__ whh_f, const bf16_t* __restrict__ whh_b,
            float* __restrict__ hs_f, float* __restrict__ hs_b) {
    const int dir = blockIdx.x;
    const float*  gx  = dir ? gx_b  : gx_f;
    const bf16_t* whh = dir ? whh_b : whh_f;
    float*        hs  = dir ? hs_b  : hs_f;

    __shared__ __align__(16) bf16_t hsh[B_ * H2_];   // 32 KB of 320 KB WGP LDS

    const int tid  = threadIdx.x;
    const int wave = tid >> 5;
    const int lane = tid & 31;
    const int lh   = lane >> 4;
    const int lm   = lane & 15;
    const int nt   = wave & 15;                // hidden-col tile 0..15
    const int mg   = wave >> 4;                // 0/1

    for (int i = tid; i < B_ * H2_; i += 1024) hsh[i] = (bf16_t)0.0f;
    __syncthreads();

    v8f c[2] = {};                             // persistent cell state tiles

    for (int ts = 0; ts < T_; ++ts) {
        const int t = dir ? (T_ - 1 - ts) : ts;
        const float* gxt = gx + ((long)(t * B_) << 10);

        // ---- issue this step's gx loads (consumed only after the k-loop)
        float gxr[2][4][8];
#pragma unroll
        for (int m = 0; m < 2; ++m) {
            const int m0 = mg * 32 + m * 16;
#pragma unroll
            for (int g = 0; g < 4; ++g) {
                const int n = g * H2_ + nt * 16 + lm;
#pragma unroll
                for (int v = 0; v < 8; ++v) {
                    int row = m0 + v + 8 * lh;
                    gxr[m][g][v] = gxt[((long)row << 10) + n];
                }
            }
        }

        // ---- prefetch next step's 256KB gx block (2 x 128B per thread)
        if (ts + 1 < T_) {
            const int tn = dir ? (T_ - 2 - ts) : (ts + 1);
            const float* gnx = gx + ((long)(tn * B_) << 10);
            __builtin_prefetch(gnx + tid * 64, 0, 0);
            __builtin_prefetch(gnx + tid * 64 + 32, 0, 0);
        }

        // ---- acc = h @ W_hh^T  (bf16 WMMA from zero, K = H2 = 256)
        v8f acc[2][4] = {};
#pragma unroll
        for (int k = 0; k < H2_; k += 32) {
            v16bf a0 = *(const v16bf*)&hsh[(mg * 32 +      lm) * H2_ + k + lh * 16];
            v16bf a1 = *(const v16bf*)&hsh[(mg * 32 + 16 + lm) * H2_ + k + lh * 16];
#pragma unroll
            for (int g = 0; g < 4; ++g) {
                const int ngl = g * H2_ + nt * 16 + lm;
                v16bf b = *(const v16bf*)(whh + (long)ngl * H2_ + k + lh * 16);
                acc[0][g] = __builtin_amdgcn_wmma_f32_16x16x32_bf16(
                                false, a0, false, b, (short)0, acc[0][g], false, false);
                acc[1][g] = __builtin_amdgcn_wmma_f32_16x16x32_bf16(
                                false, a1, false, b, (short)0, acc[1][g], false, false);
            }
        }

        // everyone must be done *reading* h before we overwrite it
        __syncthreads();

        // ---- gates + cell update, entirely in registers; write h to LDS+HBM
        const int nh = nt * 16 + lm;
#pragma unroll
        for (int m = 0; m < 2; ++m) {
            const int m0 = mg * 32 + m * 16;
#pragma unroll
            for (int v = 0; v < 8; ++v) {
                float ig = sigmoidf_(acc[m][0][v] + gxr[m][0][v]);
                float fg = sigmoidf_(acc[m][1][v] + gxr[m][1][v]);
                float gg = tanhf_   (acc[m][2][v] + gxr[m][2][v]);
                float og = sigmoidf_(acc[m][3][v] + gxr[m][3][v]);
                float cv = fg * c[m][v] + ig * gg;
                c[m][v]  = cv;
                float hv = og * tanhf_(cv);
                int row  = m0 + v + 8 * lh;                 // batch index b
                hsh[row * H2_ + nh] = (bf16_t)hv;           // for next step
                hs[((long)row * T_ + t) * H2_ + nh] = hv;   // (b,t,n) f32
            }
        }
        __syncthreads();
    }
}

// ------------------------------------------------- emissions = [hf|hb] @ W_out^T + b
__global__ __launch_bounds__(64)
void k_feats(const float* __restrict__ hs_f, const float* __restrict__ hs_b,
             const float* __restrict__ w_out, const float* __restrict__ b_out,
             float* __restrict__ emis) {
    const long bt = blockIdx.x;                 // b*T + t
    __shared__ float hrow[HID_];
    const float* hf = hs_f + bt * H2_;
    const float* hb = hs_b + bt * H2_;
    for (int i = threadIdx.x; i < H2_; i += blockDim.x) {
        hrow[i]       = hf[i];
        hrow[H2_ + i] = hb[i];
    }
    __syncthreads();
    const int k = threadIdx.x;
    if (k < K_) {
        float s = b_out[k];
        const float* w = w_out + (long)k * HID_;
#pragma unroll 8
        for (int i = 0; i < HID_; ++i) s += hrow[i] * w[i];
        emis[bt * K_ + k] = s;
    }
}

__global__ void k_zero(float* out) {
    if (threadIdx.x == 0 && blockIdx.x == 0) out[0] = 0.0f;
}

// ------------------------------------------------- CRF NLL: wave32-native scan
__global__ __launch_bounds__(1024)
void k_crf(const float* __restrict__ emis, const int* __restrict__ tags,
           const unsigned char* __restrict__ mask,
           const float* __restrict__ start, const float* __restrict__ trans,
           const float* __restrict__ endt, float* __restrict__ out) {
    __shared__ float tsh[K_ * K_];
    for (int i = threadIdx.x; i < K_ * K_; i += blockDim.x) tsh[i] = trans[i];
    __syncthreads();

    const int wave = threadIdx.x >> 5;
    const int lane = threadIdx.x & 31;
    const int b    = blockIdx.x * 32 + wave;

    const float*         em = emis + (long)b * T_ * K_;
    const int*           tg = tags + b * T_;
    const unsigned char* mk = mask + b * T_;

    // ---- gold path score (t distributed over lanes, shfl reduce)
    float emit_s = 0.0f, trans_s = 0.0f;
    int len_part = 0;
    for (int t = lane; t < T_; t += 32) {
        if (mk[t]) {
            ++len_part;
            emit_s += em[t * K_ + tg[t]];
            if (t >= 1) trans_s += tsh[tg[t - 1] * K_ + tg[t]];
        }
    }
    for (int off = 16; off; off >>= 1) {
        emit_s   += __shfl_down(emit_s, off);
        trans_s  += __shfl_down(trans_s, off);
        len_part += __shfl_down(len_part, off);
    }

    // ---- forward algorithm: alpha one tag per lane
    const int col = (lane < K_) ? lane : 0;
    float alpha = (lane < K_) ? (start[lane] + em[lane]) : -1e30f;
    for (int t = 1; t < T_; ++t) {
        float mx = -1e30f;
#pragma unroll
        for (int j = 0; j < K_; ++j) {
            float aj = __shfl(alpha, j);
            mx = fmaxf(mx, aj + tsh[j * K_ + col]);
        }
        float s = 0.0f;
#pragma unroll
        for (int j = 0; j < K_; ++j) {
            float aj = __shfl(alpha, j);
            s += __expf(aj + tsh[j * K_ + col] - mx);
        }
        float na = mx + __logf(s) + em[t * K_ + col];
        bool  mm = (mk[t] != 0);
        alpha = (lane < K_ && mm) ? na : alpha;
    }

    // ---- logZ = logsumexp(alpha + end)
    float v  = (lane < K_) ? (alpha + endt[lane]) : -1e30f;
    float mx = v;
    for (int off = 16; off; off >>= 1) mx = fmaxf(mx, __shfl_xor(mx, off));
    float s = __expf(v - mx);
    for (int off = 16; off; off >>= 1) s += __shfl_xor(s, off);
    float logZ = mx + __logf(s);

    if (lane == 0) {
        int length   = len_part;
        int last_idx = (length > 0) ? (length - 1) : 0;
        float num = start[tg[0]] + emit_s + trans_s + endt[tg[last_idx]];
        atomicAdd(out, logZ - num);
    }
}

// =================================================================== host
extern "C" void kernel_launch(void* const* d_in, const int* in_sizes, int n_in,
                              void* d_out, int out_size, void* d_ws, size_t ws_size,
                              hipStream_t stream) {
    const int*           sent   = (const int*)d_in[0];
    const int*           tags   = (const int*)d_in[1];
    const unsigned char* mask   = (const unsigned char*)d_in[2];
    const float*         emb    = (const float*)d_in[3];
    const float*         w_ih_f = (const float*)d_in[4];
    const float*         w_hh_f = (const float*)d_in[5];
    const float*         b_ih_f = (const float*)d_in[6];
    const float*         b_hh_f = (const float*)d_in[7];
    const float*         w_ih_b = (const float*)d_in[8];
    const float*         w_hh_b = (const float*)d_in[9];
    const float*         b_ih_b = (const float*)d_in[10];
    const float*         b_hh_b = (const float*)d_in[11];
    const float*         w_out  = (const float*)d_in[12];
    const float*         b_out  = (const float*)d_in[13];
    const float*         startt = (const float*)d_in[14];
    const float*         transm = (const float*)d_in[15];
    const float*         endt   = (const float*)d_in[16];
    float*               out    = (float*)d_out;

    char* ws = (char*)d_ws;
    size_t off = 0;
    auto alloc = [&](size_t bytes) -> void* {
        void* p = ws + off;
        off = (off + bytes + 255) & ~(size_t)255;
        return p;
    };
    bf16_t* xbf   = (bf16_t*)alloc((size_t)B_ * T_ * E_ * sizeof(bf16_t));
    bf16_t* wihf  = (bf16_t*)alloc((size_t)G4_ * E_ * sizeof(bf16_t));
    bf16_t* wihb  = (bf16_t*)alloc((size_t)G4_ * E_ * sizeof(bf16_t));
    bf16_t* whhf  = (bf16_t*)alloc((size_t)G4_ * H2_ * sizeof(bf16_t));
    bf16_t* whhb  = (bf16_t*)alloc((size_t)G4_ * H2_ * sizeof(bf16_t));
    float*  gxf   = (float*)alloc((size_t)B_ * T_ * G4_ * sizeof(float));
    float*  gxb   = (float*)alloc((size_t)B_ * T_ * G4_ * sizeof(float));
    float*  hsf   = (float*)alloc((size_t)B_ * T_ * H2_ * sizeof(float));
    float*  hsb   = (float*)alloc((size_t)B_ * T_ * H2_ * sizeof(float));
    float*  emis  = (float*)alloc((size_t)B_ * T_ * K_ * sizeof(float));
    (void)ws_size; (void)in_sizes; (void)n_in; (void)out_size;

    // 1) embedding gather (f32 -> bf16)
    k_embed<<<B_ * T_, E_, 0, stream>>>(sent, emb, xbf);

    // 2) weight converts to bf16
    k_cvt_bf16<<<(G4_ * E_ + 255) / 256, 256, 0, stream>>>(w_ih_f, wihf, G4_ * E_);
    k_cvt_bf16<<<(G4_ * E_ + 255) / 256, 256, 0, stream>>>(w_ih_b, wihb, G4_ * E_);
    k_cvt_bf16<<<(G4_ * H2_ + 255) / 256, 256, 0, stream>>>(w_hh_f, whhf, G4_ * H2_);
    k_cvt_bf16<<<(G4_ * H2_ + 255) / 256, 256, 0, stream>>>(w_hh_b, whhb, G4_ * H2_);

    // 3) big input-projection GEMMs (WMMA bf16, async A-tile staging in LDS)
    k_gx<<<8192, 256, 0, stream>>>(xbf, wihf, b_ih_f, b_hh_f, gxf);
    k_gx<<<8192, 256, 0, stream>>>(xbf, wihb, b_ih_b, b_hh_b, gxb);

    // 4) bidirectional LSTM scan (WMMA bf16, persistent c in VGPRs, h in LDS)
    k_lstm<<<2, 1024, 0, stream>>>(gxf, gxb, whhf, whhb, hsf, hsb);

    // 5) emission projection (K=17)
    k_feats<<<B_ * T_, 64, 0, stream>>>(hsf, hsb, w_out, b_out, emis);

    // 6) CRF negative log-likelihood
    k_zero<<<1, 1, 0, stream>>>(out);
    k_crf<<<2, 1024, 0, stream>>>(emis, tags, mask, startt, transm, endt, out);
}
